// MultiHeadAttention_9182640079581
// MI455X (gfx1250) — matrix operand order repaired and checked by reading, not compiled, not run on hardware
//
#include <hip/hip_runtime.h>

// Shapes (fixed by the reference)
// B=4, SQ=SK=1024, D_MODEL=1024, N_HEAD=16, D_K=D_V=64

typedef __attribute__((ext_vector_type(16))) _Float16 v16h;
typedef __attribute__((ext_vector_type(8)))  float    v8f;
typedef int v4i __attribute__((vector_size(16)));   // matches builtin's expected param type

#define WMMA_F16(a, b, c) \
  __builtin_amdgcn_wmma_f32_16x16x32_f16(false, (a), false, (b), (short)0, (c), false, false)

#ifndef __has_builtin
#define __has_builtin(x) 0
#endif

#if __has_builtin(__builtin_amdgcn_global_load_async_to_lds_b128)
#define USE_ASYNC_LDS 1
#else
#define USE_ASYNC_LDS 0
#endif

#if USE_ASYNC_LDS
static __device__ __forceinline__ void async_copy16(const _Float16* gsrc, _Float16* ldst) {
  __builtin_amdgcn_global_load_async_to_lds_b128(
      (__attribute__((address_space(1))) v4i*)(void*)gsrc,
      (__attribute__((address_space(3))) v4i*)(void*)ldst, 0, 0);
}
static __device__ __forceinline__ void wait_asynccnt0() {
#if __has_builtin(__builtin_amdgcn_s_wait_asynccnt)
  __builtin_amdgcn_s_wait_asynccnt(0);
#else
  asm volatile("s_wait_asynccnt 0x0" ::: "memory");
#endif
}
#endif

// ---- WMMA fragment loaders (CDNA5 wave32 layouts, CDNA5 ISA) ----
// A: 16x32 f16, source row-major [m][k] with element stride `stride`.
// lanes 0-15: m=lane, halves 0-7 -> K 0..7, halves 8-15 -> K 16..23
// lanes 16-31: m=lane-16, K 8..15 and 24..31
static __device__ __forceinline__ v16h frag_a(const _Float16* p, int stride) {
  const int lane = threadIdx.x & 31;
  const int m = lane & 15;
  const int kb = (lane >> 4) << 3; // 0 or 8
  const _Float16* q = p + m * stride + kb;
  v16h a;
#pragma unroll
  for (int i = 0; i < 8; ++i) a[i] = q[i];
#pragma unroll
  for (int i = 0; i < 8; ++i) a[8 + i] = q[16 + i];
  return a;
}

// B: 32x16 f16, source stored as [n][k] row-major (i.e. B-transposed), stride elems.
// lanes 0-15: n=lane, K 0..7 and 16..23; lanes 16-31: n=lane-16, K 8..15 and 24..31.
static __device__ __forceinline__ v16h frag_b(const _Float16* p, int stride) {
  const int lane = threadIdx.x & 31;
  const int n = lane & 15;
  const int klo = (lane >> 4) << 3; // 0 or 8
  const _Float16* q = p + n * stride + klo;
  v16h b;
#pragma unroll
  for (int i = 0; i < 8; ++i) b[i] = q[i];
#pragma unroll
  for (int i = 0; i < 8; ++i) b[8 + i] = q[16 + i];
  return b;
}

// ---------------------------------------------------------------------------
// Projection GEMM: C[4096,1024] = A[4096,1024] * W[1024,1024] (+bias)(+resid)
// 256 threads = 8 waves; WG tile 128x64; wave tile 32x32; BK=32.
// MODE 0: f16 out, scatter to [B,H,S,64]      (Q / K)
// MODE 2: f16 out, scatter to [B,H,64,S]      (V transposed)
// MODE 3: f32 out, row-major, + bias + residual (output projection)
// AF16:  A source is f16 (Xh) instead of f32 (Xf).
// ---------------------------------------------------------------------------
template <int MODE, bool AF16>
__global__ __launch_bounds__(256) void proj_gemm(
    const float* __restrict__ Xf, const _Float16* __restrict__ Xh,
    const float* __restrict__ W, const float* __restrict__ bias,
    const float* __restrict__ resid, _Float16* __restrict__ outh,
    float* __restrict__ outf)
{
  __shared__ _Float16 At[128 * 40];  // [row][k], stride 40
  __shared__ _Float16 Bt[64 * 40];   // [n][k],  stride 40
  const int t    = threadIdx.x;
  const int wid  = t >> 5;
  const int lane = t & 31;
  const int m0 = blockIdx.y * 128;
  const int n0 = blockIdx.x * 64;
  const int wm = (wid & 3) * 32;
  const int wn = (wid >> 2) * 32;

  // per-thread staging coordinates (invariant over the k-loop)
  const int arow = t >> 1;
  const int ac0  = (t & 1) * 16;
  const int bk   = t >> 3;
  const int bn0  = (t & 7) * 8;

  v8f acc[2][2] = {};

  for (int k0 = 0; k0 < 1024; k0 += 32) {
    // --- stage A tile (128x32) ---
    {
      _Float16* dst = &At[arow * 40 + ac0];
      if (AF16) {
        const _Float16* src = Xh + (size_t)(m0 + arow) * 1024 + k0 + ac0;
#if USE_ASYNC_LDS
        async_copy16(src, dst);        // 16 halves = 32B per lane, raw copy
        async_copy16(src + 8, dst + 8);
#else
#pragma unroll
        for (int j = 0; j < 16; ++j) dst[j] = src[j];
#endif
      } else {
        const float4* src = (const float4*)(Xf + (size_t)(m0 + arow) * 1024 + k0 + ac0);
#pragma unroll
        for (int j = 0; j < 4; ++j) {
          float4 v = src[j];
          dst[j * 4 + 0] = (_Float16)v.x; dst[j * 4 + 1] = (_Float16)v.y;
          dst[j * 4 + 2] = (_Float16)v.z; dst[j * 4 + 3] = (_Float16)v.w;
        }
      }
    }
    // --- stage B tile transposed: Bt[n][k] = W[k0+k][n0+n] ---
    {
      const float4* src = (const float4*)(W + (size_t)(k0 + bk) * 1024 + n0 + bn0);
      float4 v0 = src[0], v1 = src[1];
      float wv[8] = {v0.x, v0.y, v0.z, v0.w, v1.x, v1.y, v1.z, v1.w};
#pragma unroll
      for (int j = 0; j < 8; ++j) Bt[(bn0 + j) * 40 + bk] = (_Float16)wv[j];
    }
    // --- prefetch next k-step tiles into cache while WMMAs run ---
    if (k0 + 32 < 1024) {
      if (AF16)
        __builtin_prefetch(Xh + (size_t)(m0 + arow) * 1024 + (k0 + 32) + ac0, 0, 1);
      else
        __builtin_prefetch(Xf + (size_t)(m0 + arow) * 1024 + (k0 + 32) + ac0, 0, 1);
      __builtin_prefetch(W + (size_t)(k0 + 32 + bk) * 1024 + n0 + bn0, 0, 1);
    }
#if USE_ASYNC_LDS
    if (AF16) wait_asynccnt0();
#endif
    __syncthreads();

    v16h a0 = frag_a(&At[wm * 40], 40);
    v16h a1 = frag_a(&At[(wm + 16) * 40], 40);
    v16h b0 = frag_b(&Bt[wn * 40], 40);
    v16h b1 = frag_b(&Bt[(wn + 16) * 40], 40);
    acc[0][0] = WMMA_F16(a0, b0, acc[0][0]);
    acc[0][1] = WMMA_F16(a0, b1, acc[0][1]);
    acc[1][0] = WMMA_F16(a1, b0, acc[1][0]);
    acc[1][1] = WMMA_F16(a1, b1, acc[1][1]);
    __syncthreads();
  }

  // --- epilogue: C/D layout lanes0-15 -> rows r, lanes16-31 -> rows 8+r ---
  const int nl = lane & 15;
  const int rb = (lane >> 4) * 8;
#pragma unroll
  for (int i = 0; i < 2; ++i) {
#pragma unroll
    for (int j = 0; j < 2; ++j) {
      const int n = n0 + wn + j * 16 + nl;
      const float bn = bias[n];
#pragma unroll
      for (int r = 0; r < 8; ++r) {
        const int m = m0 + wm + i * 16 + rb + r;
        float v = acc[i][j][r] + bn;
        if (MODE == 3) {
          outf[(size_t)m * 1024 + n] = v + resid[(size_t)m * 1024 + n];
        } else {
          const int b = m >> 10, s = m & 1023, h = n >> 6, d = n & 63;
          size_t idx;
          if (MODE == 2) idx = (((size_t)(b * 16 + h)) * 64 + d) * 1024 + s;   // [B,H,d,S]
          else           idx = (((size_t)(b * 16 + h)) * 1024 + s) * 64 + d;   // [B,H,S,d]
          outh[idx] = (_Float16)v;
        }
      }
    }
  }
}

// ---------------------------------------------------------------------------
// Attention: one WG = 16 queries x all 1024 keys for one (b,h).
// 8 waves, wave w owns keys [w*128, w*128+128).
// Scores held in registers (8 x v8f per wave); softmax via shfl + LDS reduce;
// probs written f32 to d_out and f16 to LDS; ctx = P @ V^T, accumulated in
// LDS with ds_add_f32 atomics; ctx stored f16 merged-head [B,S,H*64].
// ---------------------------------------------------------------------------
__global__ __launch_bounds__(256) void attn_kernel(
    const _Float16* __restrict__ Qh, const _Float16* __restrict__ Kh,
    const _Float16* __restrict__ VTh, const int* __restrict__ mask,
    float* __restrict__ attn_out, _Float16* __restrict__ ctxh)
{
  __shared__ _Float16 probs[16 * 1032];  // stride 1032 halves (bank-friendly)
  __shared__ float part[16 * 64];        // ctx accumulator (atomics)
  __shared__ float redmax[16 * 8];
  __shared__ float redsum[16 * 8];

  const int t    = threadIdx.x;
  const int w    = t >> 5;
  const int lane = t & 31;
  const int nl   = lane & 15;
  const int rb   = (lane >> 4) * 8;
  const int q0   = blockIdx.x * 16;   // 64 q-blocks
  const int bh   = blockIdx.y;        // 64 (b,h) pairs
  const int b    = bh >> 4;
  const int h    = bh & 15;
  const size_t qbase = ((size_t)bh * 1024 + q0) * 64;
  const int kc0 = w * 128;

  for (int o = t; o < 1024; o += 256) part[o] = 0.0f;  // before first barrier

  // --- scores S = Q K^T / 8, masked ---
  v16h qa0 = frag_a(Qh + qbase, 64);
  v16h qa1 = frag_a(Qh + qbase + 32, 64);
  v8f s[8];
#pragma unroll
  for (int nt = 0; nt < 8; ++nt) {
    v8f acc = {};
    const size_t kb = ((size_t)bh * 1024 + kc0 + nt * 16) * 64;
    acc = WMMA_F16(qa0, frag_b(Kh + kb, 64), acc);
    acc = WMMA_F16(qa1, frag_b(Kh + kb + 32, 64), acc);
    const int mv = mask[b * 1024 + kc0 + nt * 16 + nl];
#pragma unroll
    for (int r = 0; r < 8; ++r)
      acc[r] = (mv == 0) ? -1e9f : acc[r] * 0.125f;
    s[nt] = acc;
  }

  // --- row max: per-lane over tiles, butterfly over 16 lanes, LDS over waves ---
  float rmax[8];
#pragma unroll
  for (int r = 0; r < 8; ++r) {
    float m = s[0][r];
#pragma unroll
    for (int nt = 1; nt < 8; ++nt) m = fmaxf(m, s[nt][r]);
#pragma unroll
    for (int off = 1; off < 16; off <<= 1) m = fmaxf(m, __shfl_xor(m, off, 32));
    rmax[r] = m;
  }
  if (nl == 0) {
#pragma unroll
    for (int r = 0; r < 8; ++r) redmax[(rb + r) * 8 + w] = rmax[r];
  }
  __syncthreads();
  float grmax[8];
#pragma unroll
  for (int r = 0; r < 8; ++r) {
    float m = redmax[(rb + r) * 8];
#pragma unroll
    for (int ww = 1; ww < 8; ++ww) m = fmaxf(m, redmax[(rb + r) * 8 + ww]);
    grmax[r] = m;
  }

  // --- exp + row sum ---
  float rsum[8];
#pragma unroll
  for (int r = 0; r < 8; ++r) rsum[r] = 0.0f;
#pragma unroll
  for (int nt = 0; nt < 8; ++nt) {
#pragma unroll
    for (int r = 0; r < 8; ++r) {
      float p = __expf(s[nt][r] - grmax[r]);
      s[nt][r] = p;
      rsum[r] += p;
    }
  }
#pragma unroll
  for (int r = 0; r < 8; ++r) {
#pragma unroll
    for (int off = 1; off < 16; off <<= 1) rsum[r] += __shfl_xor(rsum[r], off, 32);
  }
  if (nl == 0) {
#pragma unroll
    for (int r = 0; r < 8; ++r) redsum[(rb + r) * 8 + w] = rsum[r];
  }
  __syncthreads();
  float inv[8];
#pragma unroll
  for (int r = 0; r < 8; ++r) {
    float ss = 0.0f;
#pragma unroll
    for (int ww = 0; ww < 8; ++ww) ss += redsum[(rb + r) * 8 + ww];
    inv[r] = 1.0f / ss;
  }

  // --- normalize; write f32 probs to d_out, f16 probs to LDS ---
#pragma unroll
  for (int nt = 0; nt < 8; ++nt) {
    const int key = kc0 + nt * 16 + nl;
#pragma unroll
    for (int r = 0; r < 8; ++r) {
      float p = s[nt][r] * inv[r];
      attn_out[((size_t)bh * 1024 + q0 + rb + r) * 1024 + key] = p;
      probs[(rb + r) * 1032 + key] = (_Float16)p;
    }
  }
  __syncthreads();

  // --- ctx partial: P[16,128] @ V^T chunk -> 16x64, accumulate via LDS atomics ---
  v8f cacc[4] = {};
#pragma unroll
  for (int ks = 0; ks < 4; ++ks) {
    v16h a = frag_a(&probs[kc0 + ks * 32], 1032);
#pragma unroll
    for (int nt = 0; nt < 4; ++nt) {
      v16h bb = frag_b(VTh + ((size_t)bh * 64 + nt * 16) * 1024 + kc0 + ks * 32, 1024);
      cacc[nt] = WMMA_F16(a, bb, cacc[nt]);
    }
  }
#pragma unroll
  for (int nt = 0; nt < 4; ++nt) {
#pragma unroll
    for (int r = 0; r < 8; ++r)
      atomicAdd(&part[(rb + r) * 64 + nt * 16 + nl], cacc[nt][r]);
  }
  __syncthreads();

  // --- store ctx f16, merged-head layout [B, S, H*64] ---
  for (int o = t; o < 1024; o += 256) {
    const int row = o >> 6, col = o & 63;
    ctxh[((size_t)b * 1024 + q0 + row) * 1024 + h * 64 + col] = (_Float16)part[o];
  }
}

// ---------------------------------------------------------------------------
// LayerNorm over last dim (1024): one WG per row.
// ---------------------------------------------------------------------------
__global__ __launch_bounds__(256) void ln_kernel(
    const float* __restrict__ x, const float* __restrict__ gamma,
    const float* __restrict__ beta, float* __restrict__ out)
{
  __shared__ float rs[256], rq[256];
  const int row = blockIdx.x;
  const float* xr = x + (size_t)row * 1024;
  float v[4], sum = 0.0f, sq = 0.0f;
#pragma unroll
  for (int j = 0; j < 4; ++j) {
    v[j] = xr[threadIdx.x + j * 256];
    sum += v[j];
    sq  += v[j] * v[j];
  }
  rs[threadIdx.x] = sum;
  rq[threadIdx.x] = sq;
  __syncthreads();
  for (int off = 128; off > 0; off >>= 1) {
    if (threadIdx.x < off) {
      rs[threadIdx.x] += rs[threadIdx.x + off];
      rq[threadIdx.x] += rq[threadIdx.x + off];
    }
    __syncthreads();
  }
  const float mu  = rs[0] * (1.0f / 1024.0f);
  const float var = rq[0] * (1.0f / 1024.0f) - mu * mu;
  const float rstd = rsqrtf(var + 1e-5f);
#pragma unroll
  for (int j = 0; j < 4; ++j) {
    const int c = threadIdx.x + j * 256;
    out[(size_t)row * 1024 + c] = (v[j] - mu) * rstd * gamma[c] + beta[c];
  }
}

// ---------------------------------------------------------------------------
extern "C" void kernel_launch(void* const* d_in, const int* in_sizes, int n_in,
                              void* d_out, int out_size, void* d_ws, size_t ws_size,
                              hipStream_t stream)
{
  (void)in_sizes; (void)n_in; (void)out_size; (void)ws_size;
  const float* memory  = (const float*)d_in[0];
  const float* decoder = (const float*)d_in[1];
  const int*   mask    = (const int*)d_in[2];
  const float* Wq = (const float*)d_in[3];
  const float* bq = (const float*)d_in[4];
  const float* Wk = (const float*)d_in[5];
  const float* bk = (const float*)d_in[6];
  const float* Wv = (const float*)d_in[7];
  const float* bv = (const float*)d_in[8];
  const float* Wo = (const float*)d_in[9];
  const float* bo = (const float*)d_in[10];
  const float* gamma = (const float*)d_in[11];
  const float* beta  = (const float*)d_in[12];

  float* out  = (float*)d_out;
  float* attn = out + (size_t)4 * 1024 * 1024;  // tuple: (out, attn) concatenated

  char* ws = (char*)d_ws;                                 // 48 MB used:
  _Float16* Qh  = (_Float16*)(ws);                        //  0..8M   Q  f16 [B,H,S,64]
  _Float16* Kh  = (_Float16*)(ws + (((size_t)8)  << 20)); //  8..16M  K  f16 [B,H,S,64]
  _Float16* VTh = (_Float16*)(ws + (((size_t)16) << 20)); // 16..24M  V^T f16 [B,H,64,S]
  _Float16* Ch  = (_Float16*)(ws + (((size_t)24) << 20)); // 24..32M  ctx f16 [B,S,H*64]
  float*    Xf  = (float*)   (ws + (((size_t)32) << 20)); // 32..48M  x   f32 [4096,1024]

  const dim3 blk(256);
  const dim3 gproj(16, 32);  // N/64, M/128

  proj_gemm<0, false><<<gproj, blk, 0, stream>>>(memory,  nullptr, Wq, bq, nullptr, Qh,  nullptr);
  proj_gemm<0, false><<<gproj, blk, 0, stream>>>(memory,  nullptr, Wk, bk, nullptr, Kh,  nullptr);
  proj_gemm<2, false><<<gproj, blk, 0, stream>>>(decoder, nullptr, Wv, bv, nullptr, VTh, nullptr);
  attn_kernel<<<dim3(64, 64), blk, 0, stream>>>(Qh, Kh, VTh, mask, attn, Ch);
  proj_gemm<3, true><<<gproj, blk, 0, stream>>>(nullptr, Ch, Wo, bo, memory, nullptr, Xf);
  ln_kernel<<<dim3(4096), blk, 0, stream>>>(Xf, gamma, beta, out);
}
